// HSTULiteCTR_48352741818744
// MI455X (gfx1250) — compile-verified, matrix operand order checked by reading, not run on hardware
//
#include <hip/hip_runtime.h>
#include <hip/hip_bf16.h>
#include <math.h>

typedef __attribute__((ext_vector_type(16))) _Float16 v16h;
typedef __attribute__((ext_vector_type(8)))  float    v8f;
typedef __attribute__((ext_vector_type(2)))  _Float16 h2;

#define Bsz 4096
#define NDc 13
#define NSc 26
#define Vv  10000
#define Dd  256
#define Hh  8
#define Ll  4
#define Ss  40          // 1 + ND + NS
#define SP  48          // seq padded to 3 x 16 M-tiles
#define HDc 32
#define FF  1024
#define ATTN_SCALE 0.17677669529663687f   // 32^-0.5

__device__ __forceinline__ v8f wmma16(v16h a, v16h b, v8f c) {
  return __builtin_amdgcn_wmma_f32_16x16x32_f16(false, a, false, b, (short)0, c,
                                                false, false);
}

// Load a 16x32 f16 fragment from a row-major (A) / N-major (B) matrix.
// p = tile origin, ld = leading dimension in elements (even).
__device__ __forceinline__ v16h load_frag(const _Float16* p, int ld) {
  v16h f;
  const int lane = threadIdx.x & 31;
  const int m    = lane & 15;
  const int koff = (lane >> 4) << 3;
  const _Float16* row = p + m * ld;
#pragma unroll
  for (int i = 0; i < 8; ++i) {
    const int k = ((i & 4) << 2) + ((i & 3) << 1) + koff;
    h2 t = *(const h2*)(row + k);
    f[2 * i]     = t.x;
    f[2 * i + 1] = t.y;
  }
  return f;
}

// 3 (M) x 2 (N) register-blocked GEMM: per k-step load 3 A-frags + 2 B-frags,
// then issue 6 independent WMMAs back-to-back (fills WMMA->WMMA hazard slots).
template <int K>
__device__ __forceinline__ void gemm_block_3x2(const _Float16* __restrict__ A,
                                               int lda,
                                               const _Float16* __restrict__ Bm,
                                               int ldb, v8f acc[3][2]) {
#pragma unroll 4
  for (int kk = 0; kk < K; kk += 32) {
    v16h a0 = load_frag(A + 0 * 16 * lda + kk, lda);
    v16h a1 = load_frag(A + 1 * 16 * lda + kk, lda);
    v16h a2 = load_frag(A + 2 * 16 * lda + kk, lda);
    v16h b0 = load_frag(Bm + 0 * 16 * ldb + kk, ldb);
    v16h b1 = load_frag(Bm + 1 * 16 * ldb + kk, ldb);
    acc[0][0] = wmma16(a0, b0, acc[0][0]);
    acc[0][1] = wmma16(a0, b1, acc[0][1]);
    acc[1][0] = wmma16(a1, b0, acc[1][0]);
    acc[1][1] = wmma16(a1, b1, acc[1][1]);
    acc[2][0] = wmma16(a2, b0, acc[2][0]);
    acc[2][1] = wmma16(a2, b1, acc[2][1]);
  }
}

// ---------------------------------------------------------------------------
// Kernel 1: build x = concat([cls, dense_x @ W_dense + b, emb_lookup])  (fp32)
// ---------------------------------------------------------------------------
__global__ void build_x_kernel(const float* __restrict__ dense_x,
                               const int*   __restrict__ sparse_x,
                               const float* __restrict__ W_dense,
                               const float* __restrict__ b_dense,
                               const float* __restrict__ emb,
                               const float* __restrict__ cls,
                               float* __restrict__ x) {
  long idx = (long)blockIdx.x * blockDim.x + threadIdx.x;
  const long total = (long)Bsz * Ss * Dd;
  if (idx >= total) return;
  int  d  = (int)(idx & (Dd - 1));
  long bt = idx >> 8;
  int  t  = (int)(bt % Ss);
  int  b  = (int)(bt / Ss);
  float val;
  if (t == 0) {
    val = cls[d];
  } else if (t <= NDc) {
    int j = t - 1;
    val = b_dense[j * Dd + d];
#pragma unroll
    for (int c = 0; c < NDc; ++c)
      val += dense_x[b * NDc + c] * W_dense[c * (NDc * Dd) + j * Dd + d];
  } else {
    int s  = t - 1 - NDc;
    int id = sparse_x[b * NSc + s];
    id = id < 0 ? 0 : (id > Vv ? Vv : id);
    val = emb[((long)s * (Vv + 1) + id) * Dd + d];
  }
  x[idx] = val;
}

// ---------------------------------------------------------------------------
// Kernel 2: fp32 (KxN) -> f16 transposed (NxK) weight conversion
// ---------------------------------------------------------------------------
__global__ void convert_t_kernel(const float* __restrict__ src,
                                 _Float16* __restrict__ dst, int K, int N) {
  int idx = blockIdx.x * blockDim.x + threadIdx.x;
  if (idx >= K * N) return;
  int k = idx / N, n = idx - k * N;
  dst[n * K + k] = (_Float16)src[idx];
}

// ---------------------------------------------------------------------------
// Kernel 3: fused attention layer. One workgroup (8 waves) per batch element.
// ---------------------------------------------------------------------------
__global__ void __launch_bounds__(256, 1)
attn_kernel(float* __restrict__ x, const _Float16* __restrict__ wl,
            const float* __restrict__ bq, const float* __restrict__ bk,
            const float* __restrict__ bv, const float* __restrict__ bo,
            const float* __restrict__ g,  const float* __restrict__ beta) {
  extern __shared__ char smem[];
  float*    xs  = (float*)smem;                       // 40 x 256 fp32
  _Float16* hf  = (_Float16*)(smem + Ss * Dd * 4);    // 48 x 256
  _Float16* qf  = hf  + SP * Dd;                      // 48 x 256
  _Float16* kf  = qf  + SP * Dd;                      // 48 x 256
  _Float16* vtf = kf  + SP * Dd;                      // 256 x 64 (dim-major V^T)
  _Float16* pf  = vtf + Dd * 64;                      // 8 heads x 48 x 64
  _Float16* of  = pf  + Hh * SP * 64;                 // 48 x 256

  const int tid  = threadIdx.x;
  const int lane = tid & 31;
  const int wave = tid >> 5;
  const int lm   = lane & 15;          // C-tile column within tile
  const int mb   = (lane >> 4) << 3;   // C-tile row base (0 or 8)
  float* xrow = x + (long)blockIdx.x * Ss * Dd;

  for (int i = tid; i < Ss * Dd; i += 256) xs[i] = xrow[i];
  for (int i = tid; i < Dd * 64; i += 256) vtf[i] = (_Float16)0.f;
  for (int i = tid; i < Hh * SP * 64; i += 256) pf[i] = (_Float16)0.f;
  __syncthreads();

  // LayerNorm -> hf (f16); pad rows [40,48) with zeros
  for (int row = wave; row < Ss; row += 8) {
    const float* xr = xs + row * Dd;
    float s = 0.f, s2 = 0.f;
    for (int j = lane; j < Dd; j += 32) { float v = xr[j]; s += v; s2 += v * v; }
#pragma unroll
    for (int off = 16; off > 0; off >>= 1) {
      s  += __shfl_xor(s,  off, 32);
      s2 += __shfl_xor(s2, off, 32);
    }
    float mu   = s * (1.f / Dd);
    float rstd = rsqrtf(s2 * (1.f / Dd) - mu * mu + 1e-5f);
    for (int j = lane; j < Dd; j += 32)
      hf[row * Dd + j] = (_Float16)((xr[j] - mu) * rstd * g[j] + beta[j]);
  }
  for (int i = tid; i < (SP - Ss) * Dd; i += 256) hf[Ss * Dd + i] = (_Float16)0.f;
  __syncthreads();

  // QKV projections: each wave owns 2 N-tiles (nt0 = wave*2), all 3 M-tiles.
  const _Float16* WT[3] = { wl, wl + Dd * Dd, wl + 2 * Dd * Dd };
  const float*    BI[3] = { bq, bk, bv };
  const int nt0 = wave * 2;
#pragma unroll
  for (int j = 0; j < 3; ++j) {
    v8f acc[3][2] = {};
    __builtin_prefetch(WT[j] + nt0 * 16 * Dd, 0, 1);
    gemm_block_3x2<Dd>(hf, Dd, WT[j] + nt0 * 16 * Dd, Dd, acc);
#pragma unroll
    for (int mt = 0; mt < 3; ++mt)
#pragma unroll
      for (int jn = 0; jn < 2; ++jn) {
        int n = (nt0 + jn) * 16 + lm;
        float bn = BI[j][n];
#pragma unroll
        for (int r = 0; r < 8; ++r) {
          int m = mt * 16 + mb + r;
          _Float16 val = (_Float16)(acc[mt][jn][r] + bn);
          if (j == 0)      qf[m * Dd + n] = val;
          else if (j == 1) kf[m * Dd + n] = val;
          else             vtf[n * 64 + m] = val;   // V stored transposed
        }
      }
  }
  __syncthreads();

  // Sigmoid attention logits: one head per wave.
  // 6 fragment loads -> 9 back-to-back WMMAs; padded-key probs forced to 0.
  {
    const int head = wave;
    _Float16* pfh = pf + head * SP * 64;
    v16h a0 = load_frag(qf + 0 * 16 * Dd + head * HDc, Dd);
    v16h a1 = load_frag(qf + 1 * 16 * Dd + head * HDc, Dd);
    v16h a2 = load_frag(qf + 2 * 16 * Dd + head * HDc, Dd);
    v16h b0 = load_frag(kf + 0 * 16 * Dd + head * HDc, Dd);
    v16h b1 = load_frag(kf + 1 * 16 * Dd + head * HDc, Dd);
    v16h b2 = load_frag(kf + 2 * 16 * Dd + head * HDc, Dd);
    v8f s[3][3] = {};
    s[0][0] = wmma16(a0, b0, s[0][0]);
    s[0][1] = wmma16(a0, b1, s[0][1]);
    s[0][2] = wmma16(a0, b2, s[0][2]);
    s[1][0] = wmma16(a1, b0, s[1][0]);
    s[1][1] = wmma16(a1, b1, s[1][1]);
    s[1][2] = wmma16(a1, b2, s[1][2]);
    s[2][0] = wmma16(a2, b0, s[2][0]);
    s[2][1] = wmma16(a2, b1, s[2][1]);
    s[2][2] = wmma16(a2, b2, s[2][2]);
#pragma unroll
    for (int mt = 0; mt < 3; ++mt)
#pragma unroll
      for (int nt = 0; nt < 3; ++nt) {
        int n = nt * 16 + lm;
#pragma unroll
        for (int r = 0; r < 8; ++r) {
          float p = (n < Ss)
                        ? 1.f / (1.f + __expf(-s[mt][nt][r] * ATTN_SCALE))
                        : 0.f;
          pfh[(mt * 16 + mb + r) * 64 + n] = (_Float16)p;
        }
      }
  }
  __syncthreads();

  // o = probs @ V : per head exactly one 3x2 block (N = HD = 2 tiles), K = 64
  {
    const int head = wave;
    const _Float16* pfh = pf + head * SP * 64;
    v8f acc[3][2] = {};
    gemm_block_3x2<64>(pfh, 64, vtf + head * HDc * 64, 64, acc);
#pragma unroll
    for (int mt = 0; mt < 3; ++mt)
#pragma unroll
      for (int jn = 0; jn < 2; ++jn) {
        int n = head * HDc + jn * 16 + lm;
#pragma unroll
        for (int r = 0; r < 8; ++r)
          of[(mt * 16 + mb + r) * Dd + n] = (_Float16)acc[mt][jn][r];
      }
  }
  __syncthreads();

  // Output projection + residual -> global x  (one 3x2 block per wave)
  {
    const _Float16* WoT = wl + 3 * Dd * Dd;
    v8f acc[3][2] = {};
    __builtin_prefetch(WoT + nt0 * 16 * Dd, 0, 1);
    gemm_block_3x2<Dd>(of, Dd, WoT + nt0 * 16 * Dd, Dd, acc);
#pragma unroll
    for (int mt = 0; mt < 3; ++mt)
#pragma unroll
      for (int jn = 0; jn < 2; ++jn) {
        int n = (nt0 + jn) * 16 + lm;
        float bn = bo[n];
#pragma unroll
        for (int r = 0; r < 8; ++r) {
          int m = mt * 16 + mb + r;
          if (m < Ss) xrow[m * Dd + n] = xs[m * Dd + n] + acc[mt][jn][r] + bn;
        }
      }
  }
}

// ---------------------------------------------------------------------------
// Kernel 4: fused FFN layer. One workgroup per batch element.
// ---------------------------------------------------------------------------
__global__ void __launch_bounds__(256, 1)
ffn_kernel(float* __restrict__ x, const _Float16* __restrict__ w1t,
           const _Float16* __restrict__ w2t,
           const float* __restrict__ b1, const float* __restrict__ b2,
           const float* __restrict__ g,  const float* __restrict__ beta) {
  extern __shared__ char smem[];
  float*    xs = (float*)smem;                    // 40 x 256 fp32
  _Float16* hf = (_Float16*)(smem + Ss * Dd * 4); // 48 x 256
  _Float16* gf = hf + SP * Dd;                    // 48 x 1024

  const int tid  = threadIdx.x;
  const int lane = tid & 31;
  const int wave = tid >> 5;
  const int lm   = lane & 15;
  const int mb   = (lane >> 4) << 3;
  float* xrow = x + (long)blockIdx.x * Ss * Dd;

  for (int i = tid; i < Ss * Dd; i += 256) xs[i] = xrow[i];
  __syncthreads();

  for (int row = wave; row < Ss; row += 8) {
    const float* xr = xs + row * Dd;
    float s = 0.f, s2 = 0.f;
    for (int j = lane; j < Dd; j += 32) { float v = xr[j]; s += v; s2 += v * v; }
#pragma unroll
    for (int off = 16; off > 0; off >>= 1) {
      s  += __shfl_xor(s,  off, 32);
      s2 += __shfl_xor(s2, off, 32);
    }
    float mu   = s * (1.f / Dd);
    float rstd = rsqrtf(s2 * (1.f / Dd) - mu * mu + 1e-5f);
    for (int j = lane; j < Dd; j += 32)
      hf[row * Dd + j] = (_Float16)((xr[j] - mu) * rstd * g[j] + beta[j]);
  }
  for (int i = tid; i < (SP - Ss) * Dd; i += 256) hf[Ss * Dd + i] = (_Float16)0.f;
  __syncthreads();

  // GEMM1: 64 N-tiles -> 4 blocks of 3x2 per wave, K = 256; exact gelu -> gf
#pragma unroll
  for (int blk = 0; blk < 4; ++blk) {
    const int nt0 = wave * 8 + blk * 2;
    v8f acc[3][2] = {};
    __builtin_prefetch(w1t + nt0 * 16 * Dd, 0, 1);
    gemm_block_3x2<Dd>(hf, Dd, w1t + nt0 * 16 * Dd, Dd, acc);
#pragma unroll
    for (int mt = 0; mt < 3; ++mt)
#pragma unroll
      for (int jn = 0; jn < 2; ++jn) {
        int n = (nt0 + jn) * 16 + lm;
        float bn = b1[n];
#pragma unroll
        for (int r = 0; r < 8; ++r) {
          float v  = acc[mt][jn][r] + bn;
          float ge = 0.5f * v * (1.f + erff(v * 0.70710678f));
          gf[(mt * 16 + mb + r) * FF + n] = (_Float16)ge;
        }
      }
  }
  __syncthreads();

  // GEMM2: 16 N-tiles -> one 3x2 block per wave, K = 1024; + b2 + residual
  {
    const int nt0 = wave * 2;
    v8f acc[3][2] = {};
    __builtin_prefetch(w2t + nt0 * 16 * FF, 0, 1);
    gemm_block_3x2<FF>(gf, FF, w2t + nt0 * 16 * FF, FF, acc);
#pragma unroll
    for (int mt = 0; mt < 3; ++mt)
#pragma unroll
      for (int jn = 0; jn < 2; ++jn) {
        int n = (nt0 + jn) * 16 + lm;
        float bn = b2[n];
#pragma unroll
        for (int r = 0; r < 8; ++r) {
          int m = mt * 16 + mb + r;
          if (m < Ss) xrow[m * Dd + n] = xs[m * Dd + n] + acc[mt][jn][r] + bn;
        }
      }
  }
}

// ---------------------------------------------------------------------------
// Kernel 5: head MLP on cls token (negligible FLOPs; scalar fp32)
// ---------------------------------------------------------------------------
__global__ void head_kernel(const float* __restrict__ x,
                            const float* __restrict__ Wh1, const float* __restrict__ bh1,
                            const float* __restrict__ Wh2, const float* __restrict__ bh2,
                            const float* __restrict__ Wh3, const float* __restrict__ bh3,
                            float* __restrict__ out) {
  __shared__ float xc[256];
  __shared__ float h1[128];
  __shared__ float h2s[64];
  const int tid = threadIdx.x;
  const float* xr = x + (long)blockIdx.x * Ss * Dd;   // row 0 = cls token
  xc[tid] = xr[tid];
  xc[tid + 128] = xr[tid + 128];
  __syncthreads();
  float acc = bh1[tid];
  for (int c = 0; c < 256; ++c) acc += xc[c] * Wh1[c * 128 + tid];
  h1[tid] = fmaxf(acc, 0.f);
  __syncthreads();
  if (tid < 64) {
    float a2 = bh2[tid];
    for (int c = 0; c < 128; ++c) a2 += h1[c] * Wh2[c * 64 + tid];
    h2s[tid] = fmaxf(a2, 0.f);
  }
  __syncthreads();
  if (tid == 0) {
    float s = bh3[0];
    for (int c = 0; c < 64; ++c) s += h2s[c] * Wh3[c];
    out[blockIdx.x] = s;
  }
}

// ---------------------------------------------------------------------------
extern "C" void kernel_launch(void* const* d_in, const int* in_sizes, int n_in,
                              void* d_out, int out_size, void* d_ws, size_t ws_size,
                              hipStream_t stream) {
  const float* dense_x = (const float*)d_in[0];
  const int*   sparse_x= (const int*)  d_in[1];
  const float* W_dense = (const float*)d_in[2];
  const float* b_dense = (const float*)d_in[3];
  const float* emb     = (const float*)d_in[4];
  const float* cls     = (const float*)d_in[5];
  const float* Wq  = (const float*)d_in[6];
  const float* bq  = (const float*)d_in[7];
  const float* Wk  = (const float*)d_in[8];
  const float* bk  = (const float*)d_in[9];
  const float* Wv  = (const float*)d_in[10];
  const float* bv  = (const float*)d_in[11];
  const float* Wo  = (const float*)d_in[12];
  const float* bo  = (const float*)d_in[13];
  const float* g1  = (const float*)d_in[14];
  const float* be1 = (const float*)d_in[15];
  const float* g2  = (const float*)d_in[16];
  const float* be2 = (const float*)d_in[17];
  const float* W1  = (const float*)d_in[18];
  const float* b1  = (const float*)d_in[19];
  const float* W2  = (const float*)d_in[20];
  const float* b2  = (const float*)d_in[21];
  const float* Wh1 = (const float*)d_in[22];
  const float* bh1 = (const float*)d_in[23];
  const float* Wh2 = (const float*)d_in[24];
  const float* bh2 = (const float*)d_in[25];
  const float* Wh3 = (const float*)d_in[26];
  const float* bh3 = (const float*)d_in[27];
  float* out = (float*)d_out;

  // Workspace: [ x fp32 (B*40*256) | f16 transposed weights per layer ]
  float* x = (float*)d_ws;
  const size_t xBytes = (size_t)Bsz * Ss * Dd * sizeof(float);
  _Float16* wbuf = (_Float16*)((char*)d_ws + xBytes);
  const size_t WL = (size_t)4 * Dd * Dd + 2 * (size_t)Dd * FF;  // per-layer elems

  const long total = (long)Bsz * Ss * Dd;
  build_x_kernel<<<(int)((total + 255) / 256), 256, 0, stream>>>(
      dense_x, sparse_x, W_dense, b_dense, emb, cls, x);

  const int nqk = Dd * Dd, nf = Dd * FF;
  for (int l = 0; l < Ll; ++l) {
    _Float16* wl = wbuf + l * WL;
    convert_t_kernel<<<(nqk + 255) / 256, 256, 0, stream>>>(Wq + l * nqk, wl,            Dd, Dd);
    convert_t_kernel<<<(nqk + 255) / 256, 256, 0, stream>>>(Wk + l * nqk, wl + nqk,      Dd, Dd);
    convert_t_kernel<<<(nqk + 255) / 256, 256, 0, stream>>>(Wv + l * nqk, wl + 2 * nqk,  Dd, Dd);
    convert_t_kernel<<<(nqk + 255) / 256, 256, 0, stream>>>(Wo + l * nqk, wl + 3 * nqk,  Dd, Dd);
    convert_t_kernel<<<(nf  + 255) / 256, 256, 0, stream>>>(W1 + l * nf,  wl + 4 * nqk,      Dd, FF);
    convert_t_kernel<<<(nf  + 255) / 256, 256, 0, stream>>>(W2 + l * nf,  wl + 4 * nqk + nf, FF, Dd);
  }

  const size_t smemA = (size_t)Ss * Dd * 4 +
                       (size_t)(4 * SP * Dd + Dd * 64 + Hh * SP * 64) * 2; // 221184 B
  const size_t smemF = (size_t)Ss * Dd * 4 + (size_t)(SP * Dd + SP * FF) * 2; // 163840 B

  for (int l = 0; l < Ll; ++l) {
    _Float16* wl = wbuf + l * WL;
    attn_kernel<<<Bsz, 256, smemA, stream>>>(x, wl,
        bq + l * Dd, bk + l * Dd, bv + l * Dd, bo + l * Dd,
        g1 + l * Dd, be1 + l * Dd);
    ffn_kernel<<<Bsz, 256, smemF, stream>>>(x,
        wl + 4 * nqk, wl + 4 * nqk + nf,
        b1 + l * FF, b2 + l * Dd, g2 + l * Dd, be2 + l * Dd);
  }

  head_kernel<<<Bsz, 128, 0, stream>>>(x, Wh1, bh1, Wh2, bh2, Wh3, bh3, out);
}
